// TropicalCausalSelfAttention_74096775790957
// MI455X (gfx1250) — matrix operand order repaired and checked by reading, not compiled, not run on hardware
//
#include <hip/hip_runtime.h>
#include <hip/hip_bf16.h>
#include <stdint.h>

// Problem constants from the reference: B=1, T=512, C=512, H=8, D=64
#define T_ 512
#define C_ 512
#define H_ 8
#define D_ 64

typedef __attribute__((ext_vector_type(16))) __bf16        v16bf;
typedef __attribute__((ext_vector_type(8)))  float         v8f;
typedef __attribute__((ext_vector_type(4)))  unsigned int  u32x4;
typedef int v4i __attribute__((vector_size(16)));   // matches builtin param type

#define AS1 __attribute__((address_space(1)))
#define AS3 __attribute__((address_space(3)))

#if __has_builtin(__builtin_amdgcn_global_load_async_to_lds_b128)
#define HAVE_ASYNC_LDS 1
#else
#define HAVE_ASYNC_LDS 0
#endif

#if __has_builtin(__builtin_amdgcn_s_wait_asynccnt)
#define WAIT_ASYNC() __builtin_amdgcn_s_wait_asynccnt(0)
#else
#define WAIT_ASYNC() asm volatile("s_wait_asynccnt 0x0" ::: "memory")
#endif

union FragBF {
  u32x4 u[2];
  v16bf v;
};

__device__ __forceinline__ unsigned short f2bf_bits(float f) {
  // round-to-nearest-even f32 -> bf16 (bit pattern)
  unsigned int u = __float_as_uint(f);
  unsigned int r = u + 0x7FFFu + ((u >> 16) & 1u);
  return (unsigned short)(r >> 16);
}

// ---------------------------------------------------------------------------
// f32 -> bf16 elementwise convert
// ---------------------------------------------------------------------------
__global__ void cvt_bf16_kernel(const float* __restrict__ in,
                                unsigned short* __restrict__ out, int n) {
  int i = blockIdx.x * blockDim.x + threadIdx.x;
  if (i < n) out[i] = f2bf_bits(in[i]);
}

// ---------------------------------------------------------------------------
// W (512x512 row-major f32, K x N) -> W^T (N x K row-major bf16)
// ---------------------------------------------------------------------------
__global__ void cvt_transpose_kernel(const float* __restrict__ in,
                                     unsigned short* __restrict__ out) {
  int i = blockIdx.x * blockDim.x + threadIdx.x;   // over 512*512
  int k = i >> 9, n = i & 511;
  out[n * C_ + k] = f2bf_bits(in[i]);
}

// ---------------------------------------------------------------------------
// Batched GEMM: C[M,N] (f32) = A[M,K] (bf16 row-major) * Bt[N,K]^T (bf16
// row-major N x K).  One wave computes a 16x64 strip: the A fragment is
// loaded once per K-step and reused by FOUR v_wmma_f32_16x16x32_bf16
// issues (4x arithmetic intensity vs one-tile-per-wave).
//
// Fragment layouts per CDNA5 ISA 7.12.2 (wave32):
//   A 16x32:  lane (r,h): row r, VGPR0-3 -> K = 8h..8h+7, VGPR4-7 -> K=16+8h..
//   B 32x16:  lane (r,h): col r, VGPR0-3 -> K = 8h..8h+7, VGPR4-7 -> K=16+8h..
//   C/D:      lane (r,h): col r, VGPR g -> row g + 8h
// ---------------------------------------------------------------------------
__global__ void gemm_bf16_tn(const unsigned short* __restrict__ A,
                             const unsigned short* __restrict__ Bt,
                             float* __restrict__ Cc,
                             int M, int N, int K,
                             int lda, int ldb, int ldc,
                             long long sA, long long sB, long long sC) {
  int b = blockIdx.y;
  A  += (long long)b * sA;
  Bt += (long long)b * sB;
  Cc += (long long)b * sC;

  int lane = threadIdx.x & 31;
  int wid  = threadIdx.x >> 5;
  int gw   = blockIdx.x * (blockDim.x >> 5) + wid;
  int ntn4 = N >> 6;                       // strips of 64 columns
  int tm   = gw / ntn4;
  int tn4  = gw - tm * ntn4;
  if (tm >= (M >> 4)) return;

  int r  = lane & 15;
  int hh = lane >> 4;

  const unsigned short* arow = A  + (long long)((tm << 4) + r) * lda + 8 * hh;
  const unsigned short* b0   = Bt + (long long)((tn4 << 6) + r) * ldb + 8 * hh;

  v8f acc[4];
#pragma unroll
  for (int t = 0; t < 4; ++t) acc[t] = (v8f)0.0f;

  for (int k0 = 0; k0 < K; k0 += 32) {
    FragBF fa;
    const u32x4* pa = (const u32x4*)(arow + k0);
    fa.u[0] = pa[0];            // K = k0+8h    .. +7
    fa.u[1] = pa[2];            // K = k0+8h+16 .. +7  (+16 elements)
#pragma unroll
    for (int t = 0; t < 4; ++t) {
      FragBF fb;
      const u32x4* pb = (const u32x4*)(b0 + (long long)(t * 16) * ldb + k0);
      fb.u[0] = pb[0];          // K = k0+8h    .. +7
      fb.u[1] = pb[2];          // K = k0+8h+16 .. +7  (+16 elements)
      acc[t] = __builtin_amdgcn_wmma_f32_16x16x32_bf16(
          /*neg_a=*/false, fa.v, /*neg_b=*/false, fb.v,
          /*c_mod=*/(short)0, acc[t], /*reuse_a=*/false, /*reuse_b=*/false);
    }
  }

#pragma unroll
  for (int t = 0; t < 4; ++t) {
    float* cp = Cc + (long long)((tm << 4) + (hh << 3)) * ldc +
                (tn4 << 6) + t * 16 + r;
#pragma unroll
    for (int g = 0; g < 8; ++g) cp[(long long)g * ldc] = acc[t][g];
  }
}

// ---------------------------------------------------------------------------
// Fused RoPE + RMS-norm, in place on q and k ((T, H*D) f32).
// One wave per (tensor, t, h); lane d in [0,32) handles the pair (d, d+32).
// ---------------------------------------------------------------------------
__global__ void rope_rms_kernel(float* __restrict__ q, float* __restrict__ kbuf,
                                const float* __restrict__ cosp,
                                const float* __restrict__ sinp) {
  int w    = blockIdx.x * (blockDim.x >> 5) + (threadIdx.x >> 5);
  int lane = threadIdx.x & 31;
  float* buf = (w < T_ * H_) ? q : kbuf;
  int idx = w & (T_ * H_ - 1);
  int t = idx >> 3, h = idx & 7;
  float* p = buf + (long long)t * (H_ * D_) + h * D_;

  float x1 = p[lane];
  float x2 = p[lane + 32];
  float c  = cosp[t * (D_ / 2) + lane];
  float s  = sinp[t * (D_ / 2) + lane];
  float y1 =  x1 * c + x2 * s;
  float y2 = -x1 * s + x2 * c;

  float ss = y1 * y1 + y2 * y2;
#pragma unroll
  for (int off = 16; off > 0; off >>= 1) ss += __shfl_xor(ss, off, 32);
  float scale = rsqrtf(ss * (1.0f / 64.0f) + 1e-6f);

  p[lane]      = y1 * scale;
  p[lane + 32] = y2 * scale;
}

// ---------------------------------------------------------------------------
// Fused tropical scores + causal mask + softmax -> bf16 weights.
// One block (256 threads) per (qt, h) row.
//
// k is consumed in 64x64 f32 tiles staged into LDS with
// GLOBAL_LOAD_ASYNC_TO_LDS_B128 (ASYNCcnt + s_wait_asynccnt), then the
// tropical scores[kk] = max_d(q[d] + k[kk,d]) are computed from LDS with a
// 4-way split over d per column.  Row max / sum via LDS tree reductions.
// Emits w = softmax row in bf16 (H x T x T row-major, the WMMA A-operand
// of the w@v GEMM).
// ---------------------------------------------------------------------------
__global__ void scores_softmax_kernel(const float* __restrict__ q,
                                      const float* __restrict__ k,
                                      unsigned short* __restrict__ wb) {
  int qt = blockIdx.x, h = blockIdx.y;
  __shared__ float qs[D_];
  __shared__ float ks[64 * D_];      // 16 KB k-tile
  __shared__ float srow[T_];         // tropical scores for this row
  __shared__ float red[256];
  __shared__ float red2[4 * 64];
  int tid = threadIdx.x;

  if (tid < D_) qs[tid] = q[(long long)qt * (H_ * D_) + h * D_ + tid];
  srow[tid]       = -3.4e38f;
  srow[tid + 256] = -3.4e38f;
  __syncthreads();

  int ntile = (qt >> 6) + 1;         // causal: only tiles with base <= qt
  for (int kt = 0; kt < ntile; ++kt) {
    // ---- stage 64 rows x 64 d of k (head h) into LDS -------------------
#if HAVE_ASYNC_LDS
#pragma unroll
    for (int i = 0; i < 4; ++i) {
      int j  = tid + i * 256;        // 16-byte chunk id, 1024 total
      int rr = j >> 4;               // row within tile
      int d4 = (j & 15) << 2;        // starting d
      const float* src =
          k + (long long)(kt * 64 + rr) * (H_ * D_) + h * D_ + d4;
      float* dst = &ks[rr * D_ + d4];
      __builtin_amdgcn_global_load_async_to_lds_b128(
          (AS1 v4i*)(uintptr_t)src, (AS3 v4i*)(uintptr_t)dst, 0, 0);
    }
    WAIT_ASYNC();
#else
#pragma unroll
    for (int i = 0; i < 4; ++i) {
      int j  = tid + i * 256;
      int rr = j >> 4;
      int d4 = (j & 15) << 2;
      const float4* src = (const float4*)(
          k + (long long)(kt * 64 + rr) * (H_ * D_) + h * D_ + d4);
      *(float4*)&ks[rr * D_ + d4] = *src;
    }
#endif
    __syncthreads();

    // ---- tropical max-plus: 4 threads per column, 16 d each ------------
    int col = tid & 63;
    int grp = tid >> 6;
    const float* kc = &ks[col * D_ + grp * 16];
    const float* qc = &qs[grp * 16];
    float m = -3.4e38f;
#pragma unroll
    for (int d = 0; d < 16; ++d) m = fmaxf(m, qc[d] + kc[d]);
    red2[grp * 64 + col] = m;
    __syncthreads();

    if (tid < 64) {
      float mm = fmaxf(fmaxf(red2[tid], red2[64 + tid]),
                       fmaxf(red2[128 + tid], red2[192 + tid]));
      int kk = kt * 64 + tid;
      srow[kk] = (kk <= qt) ? mm : -3.4e38f;
    }
    __syncthreads();                 // protect ks/red2 before next tile
  }

  // ---- softmax over srow[0..511] --------------------------------------
  float sc0 = srow[tid];
  float sc1 = srow[tid + 256];
  red[tid] = fmaxf(sc0, sc1);
  __syncthreads();
#pragma unroll
  for (int s = 128; s > 0; s >>= 1) {
    if (tid < s) red[tid] = fmaxf(red[tid], red[tid + s]);
    __syncthreads();
  }
  float rowmax = red[0];
  __syncthreads();

  float e0 = (tid <= qt)       ? __expf(sc0 - rowmax) : 0.f;
  float e1 = (tid + 256 <= qt) ? __expf(sc1 - rowmax) : 0.f;
  red[tid] = e0 + e1;
  __syncthreads();
#pragma unroll
  for (int s = 128; s > 0; s >>= 1) {
    if (tid < s) red[tid] += red[tid + s];
    __syncthreads();
  }
  float inv = 1.0f / red[0];

  unsigned short* wrow = wb + ((long long)h * T_ + qt) * T_;
  wrow[tid]       = f2bf_bits(e0 * inv);
  wrow[tid + 256] = f2bf_bits(e1 * inv);
}

// ---------------------------------------------------------------------------
// v (T, H*D f32) -> per-head transposed bf16: vT[h][d][t]  (B^T for w@v GEMM)
// ---------------------------------------------------------------------------
__global__ void vT_kernel(const float* __restrict__ v,
                          unsigned short* __restrict__ vT) {
  int i = blockIdx.x * blockDim.x + threadIdx.x;  // over T*H*D
  int t = i >> 9, c = i & 511;
  int h = c >> 6, d = c & 63;
  vT[(long long)h * (D_ * T_) + d * T_ + t] = f2bf_bits(v[i]);
}

// ---------------------------------------------------------------------------
// Host-side orchestration
// ---------------------------------------------------------------------------
extern "C" void kernel_launch(void* const* d_in, const int* in_sizes, int n_in,
                              void* d_out, int out_size, void* d_ws,
                              size_t ws_size, hipStream_t stream) {
  (void)in_sizes; (void)n_in; (void)out_size; (void)ws_size;

  const float* x    = (const float*)d_in[0];
  const float* cosp = (const float*)d_in[1];
  const float* sinp = (const float*)d_in[2];
  const float* Wq   = (const float*)d_in[3];
  const float* Wk   = (const float*)d_in[4];
  const float* Wv   = (const float*)d_in[5];
  const float* Wo   = (const float*)d_in[6];
  float* out = (float*)d_out;

  char* ws = (char*)d_ws;
  size_t off = 0;
  auto alloc = [&](size_t bytes) -> char* {
    char* p = ws + off;
    off += (bytes + 255) & ~(size_t)255;
    return p;
  };

  unsigned short* xb  = (unsigned short*)alloc((size_t)T_ * C_ * 2);
  unsigned short* WqT = (unsigned short*)alloc((size_t)C_ * C_ * 2);
  unsigned short* WkT = (unsigned short*)alloc((size_t)C_ * C_ * 2);
  unsigned short* WvT = (unsigned short*)alloc((size_t)C_ * C_ * 2);
  unsigned short* WoT = (unsigned short*)alloc((size_t)C_ * C_ * 2);
  float* qf = (float*)alloc((size_t)T_ * C_ * 4);
  float* kf = (float*)alloc((size_t)T_ * C_ * 4);
  float* vf = (float*)alloc((size_t)T_ * C_ * 4);
  unsigned short* wb = (unsigned short*)alloc((size_t)H_ * T_ * T_ * 2);
  unsigned short* vT = (unsigned short*)alloc((size_t)H_ * D_ * T_ * 2);
  float* yf = (float*)alloc((size_t)T_ * C_ * 4);
  unsigned short* yb = (unsigned short*)alloc((size_t)T_ * C_ * 2);

  const int BLK = 256;
  const int NELEM = T_ * C_;                     // 262144
  const int CVTG = NELEM / BLK;                  // 1024

  // 1) downconvert activations and pre-transpose all weight matrices to bf16
  cvt_bf16_kernel<<<CVTG, BLK, 0, stream>>>(x, xb, NELEM);
  cvt_transpose_kernel<<<CVTG, BLK, 0, stream>>>(Wq, WqT);
  cvt_transpose_kernel<<<CVTG, BLK, 0, stream>>>(Wk, WkT);
  cvt_transpose_kernel<<<CVTG, BLK, 0, stream>>>(Wv, WvT);
  cvt_transpose_kernel<<<CVTG, BLK, 0, stream>>>(Wo, WoT);

  // 2) q,k,v = x @ W  (512x512x512, WMMA bf16 -> f32)
  //    16x64 strips: 32 * 8 = 256 waves -> 32 blocks of 8 waves
  gemm_bf16_tn<<<dim3(32, 1), BLK, 0, stream>>>(xb, WqT, qf, 512, 512, 512,
                                                512, 512, 512, 0, 0, 0);
  gemm_bf16_tn<<<dim3(32, 1), BLK, 0, stream>>>(xb, WkT, kf, 512, 512, 512,
                                                512, 512, 512, 0, 0, 0);
  gemm_bf16_tn<<<dim3(32, 1), BLK, 0, stream>>>(xb, WvT, vf, 512, 512, 512,
                                                512, 512, 512, 0, 0, 0);

  // 3) RoPE + RMS-norm in place on q and k: 2*T*H = 8192 waves / 8 = 1024
  rope_rms_kernel<<<2 * T_ * H_ / 8, BLK, 0, stream>>>(qf, kf, cosp, sinp);

  // 4) tropical max-plus scores + causal softmax -> bf16 attention weights
  scores_softmax_kernel<<<dim3(T_, H_), BLK, 0, stream>>>(qf, kf, wb);

  // 5) per-head v transpose to bf16
  vT_kernel<<<CVTG, BLK, 0, stream>>>(vf, vT);

  // 6) y[h] = w[h] @ v[h]  (batched 512x64x512; C batch stride = 64 columns)
  //    16x64 strips: 32 waves per head -> 4 blocks, 8 head-batches
  gemm_bf16_tn<<<dim3(4, H_), BLK, 0, stream>>>(
      wb, vT, yf, 512, 64, 512, 512, 512, 512,
      (long long)T_ * T_, (long long)D_ * T_, (long long)D_);

  // 7) out = y @ Wo
  cvt_bf16_kernel<<<CVTG, BLK, 0, stream>>>(yf, yb, NELEM);
  gemm_bf16_tn<<<dim3(32, 1), BLK, 0, stream>>>(yb, WoT, out, 512, 512, 512,
                                                512, 512, 512, 0, 0, 0);
}
